// GMMBlock_20822001451603
// MI455X (gfx1250) — matrix operand order repaired and checked by reading, not compiled, not run on hardware
//
#include <hip/hip_runtime.h>
#include <hip/hip_bf16.h>

typedef __attribute__((ext_vector_type(16))) __bf16 v16bf;
typedef __attribute__((ext_vector_type(8)))  float  v8f;

#define Bn   16
#define Cn   16
#define Hn   128
#define Wn   128
#define Kn   8
#define HW   (Hn*Wn)            // 16384
#define NPIX (Bn*HW)            // 262144
#define CIN  152                // C + C*K + K
#define COUT 128                // C*K
#define NT   8                  // COUT / 16
#define KT   5                  // CIN padded to 160 = 5 k-tiles of 32

// workspace layout (bytes): only packed weights now (rho stays on-chip)
#define WMU_FRAG_OFF  0         // bf16, NT*KT*32*16 = 20480 elems (40960 B)
#define WSIG_FRAG_OFF 40960     // bf16, KT*32*16 = 2560 elems (5120 B)

#define INV_2PI 0.15915494309189535f

// ---------------------------------------------------------------------------
// Kernel 1: repack Wmu [128,152] f32 and Wsig [8,152] f32 into bf16 WMMA
// B fragments. B layout (16-bit B, 32x16): lane<16 -> N=lane, elements hold
// K = e (e<8) or e+8 (e>=8) within a 32-wide k-tile; lanes>=16 mirror with K +8.
// K padded 152->160 with zeros; Wsig N padded 8->16 with zeros.
// ---------------------------------------------------------------------------
__global__ void pack_weights_kernel(const float* __restrict__ Wmu,
                                    const float* __restrict__ Wsig,
                                    __bf16* __restrict__ wmu_f,
                                    __bf16* __restrict__ wsig_f) {
    int tid = blockIdx.x * 256 + threadIdx.x;
    const int per = KT * 32 * 16;                 // 2560 elems per n-tile
    if (tid >= (NT + 1) * per) return;
    int nt  = tid / per;
    int rem = tid % per;
    int kt  = rem / 512;
    int l   = (rem % 512) / 16;
    int e   = rem % 16;
    int half = l >> 4;
    int Kg = kt * 32 + ((e < 8) ? (half * 8 + e) : (16 + half * 8 + (e - 8)));             // global K (channel) index
    float v = 0.0f;
    if (nt < NT) {
        int col = nt * 16 + (l & 15);             // Cout
        if (Kg < CIN) v = Wmu[col * CIN + Kg];
        wmu_f[(nt * KT + kt) * 512 + l * 16 + e] = (__bf16)v;
    } else {
        int col = l & 15;
        if (col < Kn && Kg < CIN) v = Wsig[col * CIN + Kg];
        wsig_f[kt * 512 + l * 16 + e] = (__bf16)v;
    }
}

// ---------------------------------------------------------------------------
// Kernel 2 (fused): one wave32 per 16-pixel tile.
//  Phase 1 (VALU): per lane handles (pixel m = lane&15, k = half*4 + 0..3):
//    dens, alpha, rho, and pi_new = softmax(conv 16->8) via one shfl_xor(16).
//  Phase 2: build A = [x | mu | rho | 0] fragments (rho straight from
//    registers, no memory round-trip), stash fragment-ordered copy in LDS.
//  Phase 3: mu_new = relu(A @ Wmu^T + bmu), 8 N-tiles x 5 K-tiles of WMMA;
//    scatter bf16 mu_new into the LDS A2 slots (D-layout -> A-layout).
//  Phase 4: sigma_new = exp(relu(A2 @ Wsig^T + bsig)), 5 more WMMAs.
// A fragment layout (16-bit A 16x32): lane<16 -> M=lane, K {0-7,16-23};
// lanes>=16 -> same M, K {8-15,24-31}.  D: lane_lo=N, VGPR r -> M=r+half*8.
// ---------------------------------------------------------------------------
__global__ void __launch_bounds__(128)
fused_wmma_kernel(const float* __restrict__ x,
                  const float* __restrict__ pi,
                  const float* __restrict__ mu,
                  const float* __restrict__ sigma,
                  const float* __restrict__ Wpi,
                  const float* __restrict__ bpi,
                  const __bf16* __restrict__ wmu_f,
                  const __bf16* __restrict__ wsig_f,
                  const float* __restrict__ bmu,
                  const float* __restrict__ bsig,
                  float* __restrict__ pi_new,
                  float* __restrict__ mu_new,
                  float* __restrict__ sigma_new) {
    __shared__ __align__(32) char smem[4 * KT * 1024];   // 4 waves * 5 KB

    int wave    = threadIdx.x >> 5;
    int lane    = threadIdx.x & 31;
    int m       = lane & 15;                 // pixel row (A: M, D: N index)
    int half    = lane >> 4;
    int tile    = blockIdx.x * 4 + wave;

    int p_base  = tile * 16;                 // 16 consecutive flat pixels, one b
    int b       = p_base / HW;
    int sp_base = p_base % HW;
    int sp_m    = sp_base + m;               // this lane's pixel

    char* lds = smem + wave * (KT * 1024);
    const float* xb  = x  + (size_t)b * Cn   * HW;
    const float* mub = mu + (size_t)b * COUT * HW;

    // ---------------- Phase 1: density / alpha / rho / pi_new --------------
    float xv[Cn];
#pragma unroll
    for (int c = 0; c < Cn; ++c) xv[c] = xb[c * HW + sp_m];

    float piv[4], alphav[4], rhov[4];
#pragma unroll
    for (int j = 0; j < 4; ++j) {
        int k = half * 4 + j;
        float dist = 0.0f;
#pragma unroll
        for (int c = 0; c < Cn; ++c) {
            float d = xv[c] - mub[(k * Cn + c) * HW + sp_m];
            dist += d * d;
        }
        float s  = sigma[(size_t)b * Kn * HW + k * HW + sp_m];
        float s2 = s * s;
        float invs2 = 1.0f / s2;
        float t  = invs2 * INV_2PI;                    // 1/(2 pi s^2)
        float t2 = t * t, t4 = t2 * t2, t8 = t4 * t4;  // ^(C/2) = ^8
        float dens = t8 * expf(-0.5f * dist * invs2);
        piv[j]    = pi[(size_t)b * Kn * HW + k * HW + sp_m];
        alphav[j] = piv[j] * dens;
        rhov[j]   = alphav[j] * dens;
    }

    // pi_new: each half holds 4 of the 8 input (pi,alpha) pairs; combine
    // partial dot-products across the half boundary with shfl_xor(16).
    float zp[Kn], zmax = -3.0e38f;
#pragma unroll
    for (int o = 0; o < Kn; ++o) {
        float part = 0.0f;
#pragma unroll
        for (int j = 0; j < 4; ++j) {
            int kj = half * 4 + j;
            part += Wpi[o * 2 * Kn + kj] * piv[j]
                  + Wpi[o * 2 * Kn + Kn + kj] * alphav[j];
        }
        part += __shfl_xor(part, 16, 32);
        zp[o] = part + bpi[o];
        zmax  = fmaxf(zmax, zp[o]);
    }
    float ssum = 0.0f;
#pragma unroll
    for (int o = 0; o < Kn; ++o) { zp[o] = expf(zp[o] - zmax); ssum += zp[o]; }
    float rs = 1.0f / ssum;
#pragma unroll
    for (int j = 0; j < 4; ++j) {
        int k = half * 4 + j;
        float zsel = half ? zp[4 + j] : zp[j];          // compile-time indices
        pi_new[(size_t)b * Kn * HW + k * HW + sp_m] = zsel * rs;
    }

    // bring the other half's rho over: half-0 lanes need all 8 rho values
    float rho_other[4];
#pragma unroll
    for (int j = 0; j < 4; ++j) rho_other[j] = __shfl_xor(rhov[j], 16, 32);

    // ---------------- Phase 2: build A fragments + LDS copy ----------------
    v16bf afrag[KT];
#pragma unroll
    for (int kt = 0; kt < KT; ++kt) {
#pragma unroll
        for (int e = 0; e < 16; ++e) {
            int klc = (e < 8) ? e : e + 8;          // half-0 K within tile
            int ch0 = kt * 32 + klc;                // half-0 channel (constexpr)
            float v;
            if (ch0 < Cn) {                         // x region (kt=0, e<8)
                v = half ? xv[ch0 + 8] : xv[ch0];
            } else if (ch0 < Cn + COUT) {           // mu region
                v = mub[(ch0 - Cn + half * 8) * HW + sp_m];
            } else if (ch0 < CIN) {                 // rho (half0) / pad (half1)
                int kk = ch0 - (Cn + COUT);         // 0..7, constexpr
                float r = (kk < 4) ? rhov[kk] : rho_other[kk - 4];
                v = half ? 0.0f : r;
            } else {
                v = 0.0f;                           // pad
            }
            afrag[kt][e] = (__bf16)v;
        }
        v16bf a = afrag[kt];
        uint4* dst = reinterpret_cast<uint4*>(lds + kt * 1024 + lane * 32);
        const uint4* src = reinterpret_cast<const uint4*>(&a);
        dst[0] = src[0];
        dst[1] = src[1];
    }

    // ---------------- Phase 3: mu_new GEMM (8 N-tiles x 5 K-tiles) ---------
#pragma unroll
    for (int nt = 0; nt < NT; ++nt) {
        v8f acc = {};
#pragma unroll
        for (int kt = 0; kt < KT; ++kt) {
            v16bf bf = *reinterpret_cast<const v16bf*>(
                wmu_f + ((size_t)(nt * KT + kt) * 32 + lane) * 16);
            acc = __builtin_amdgcn_wmma_f32_16x16x32_bf16(
                false, afrag[kt], false, bf, (short)0, acc, false, false);
        }
        int ch_out = nt * 16 + m;                // D: N = lane&15
        float bias = bmu[ch_out];
#pragma unroll
        for (int r = 0; r < 8; ++r) {
            int mrow = r + half * 8;             // D: M
            float v = fmaxf(acc[r] + bias, 0.0f);
            mu_new[(size_t)b * COUT * HW + ch_out * HW + sp_base + mrow] = v;
            // scatter bf16 into LDS A2 slot (transpose D-layout -> A-layout)
            int ch2 = Cn + ch_out;
            int kt2 = ch2 >> 5, r32 = ch2 & 31;
            int h2  = (r32 >> 3) & 1;
            int e2  = (r32 < 16) ? (r32 & 7) : (8 + (r32 & 7));
            int tl  = h2 * 16 + mrow;
            *reinterpret_cast<__bf16*>(lds + kt2 * 1024 + tl * 32 + e2 * 2) = (__bf16)v;
        }
    }

    __syncthreads();

    // ---------------- Phase 4: sigma GEMM (N padded to 16) -----------------
    v8f acc2 = {};
#pragma unroll
    for (int kt = 0; kt < KT; ++kt) {
        v16bf a2 = *reinterpret_cast<const v16bf*>(lds + kt * 1024 + lane * 32);
        v16bf bf = *reinterpret_cast<const v16bf*>(wsig_f + ((size_t)(kt * 32 + lane)) * 16);
        acc2 = __builtin_amdgcn_wmma_f32_16x16x32_bf16(
            false, a2, false, bf, (short)0, acc2, false, false);
    }
    if (m < Kn) {                                // first 8 N are real outputs
        float bias = bsig[m];
#pragma unroll
        for (int r = 0; r < 8; ++r) {
            int mrow = r + half * 8;
            float v = expf(fmaxf(acc2[r] + bias, 0.0f));
            sigma_new[(size_t)b * Kn * HW + m * HW + sp_base + mrow] = v;
        }
    }
}

// ---------------------------------------------------------------------------
// Kernel 3: dens2 + gamma = sigmoid(conv1x1(pi_new * dens2, Wg) + bg).
// mu_new / sigma_new / pi_new are hot in the 192 MB L2 at this point.
// ---------------------------------------------------------------------------
__global__ void dens2_gamma_kernel(const float* __restrict__ x,
                                   const float* __restrict__ mu_new,
                                   const float* __restrict__ sigma_new,
                                   const float* __restrict__ pi_new,
                                   const float* __restrict__ Wg,
                                   const float* __restrict__ bg,
                                   float* __restrict__ gamma) {
    int p = blockIdx.x * 256 + threadIdx.x;
    if (p >= NPIX) return;
    int b = p / HW, sp = p % HW;

    const float* xb = x + (size_t)b * Cn * HW + sp;
    float xv[Cn];
#pragma unroll
    for (int c = 0; c < Cn; ++c) xv[c] = xb[c * HW];

    const float* mb = mu_new + (size_t)b * COUT * HW + sp;
    float acc = 0.0f;
#pragma unroll
    for (int k = 0; k < Kn; ++k) {
        float dist = 0.0f;
#pragma unroll
        for (int c = 0; c < Cn; ++c) {
            float d = xv[c] - mb[(k * Cn + c) * HW];
            dist += d * d;
        }
        float s  = sigma_new[(size_t)b * Kn * HW + k * HW + sp];
        float s2 = s * s;
        float invs2 = 1.0f / s2;
        float t  = invs2 * INV_2PI;
        float t2 = t * t, t4 = t2 * t2, t8 = t4 * t4;
        float dens2 = t8 * expf(-0.5f * dist * invs2);
        acc += Wg[k] * pi_new[(size_t)b * Kn * HW + k * HW + sp] * dens2;
    }
    float g = acc + bg[0];
    gamma[p] = 1.0f / (1.0f + expf(-g));
}

// ---------------------------------------------------------------------------
extern "C" void kernel_launch(void* const* d_in, const int* in_sizes, int n_in,
                              void* d_out, int out_size, void* d_ws, size_t ws_size,
                              hipStream_t stream) {
    const float* x     = (const float*)d_in[0];
    const float* pi    = (const float*)d_in[1];
    const float* mu    = (const float*)d_in[2];
    const float* sigma = (const float*)d_in[3];
    const float* Wpi   = (const float*)d_in[4];
    const float* bpi   = (const float*)d_in[5];
    const float* Wmu   = (const float*)d_in[6];
    const float* bmu   = (const float*)d_in[7];
    const float* Wsig  = (const float*)d_in[8];
    const float* bsig  = (const float*)d_in[9];
    const float* Wg    = (const float*)d_in[10];
    const float* bg    = (const float*)d_in[11];

    float* out       = (float*)d_out;
    float* pi_new    = out;                                    // [B,K,H,W]
    float* mu_new    = pi_new + (size_t)Bn * Kn * HW;          // [B,C*K,H,W]
    float* sigma_new = mu_new + (size_t)Bn * COUT * HW;        // [B,K,H,W]
    float* gamma     = sigma_new + (size_t)Bn * Kn * HW;       // [B,1,H,W]

    char*   ws     = (char*)d_ws;
    __bf16* wmu_f  = (__bf16*)(ws + WMU_FRAG_OFF);
    __bf16* wsig_f = (__bf16*)(ws + WSIG_FRAG_OFF);

    pack_weights_kernel<<<((NT + 1) * KT * 512 + 255) / 256, 256, 0, stream>>>(
        Wmu, Wsig, wmu_f, wsig_f);

    fused_wmma_kernel<<<(NPIX / 16) / 4, 128, 0, stream>>>(
        x, pi, mu, sigma, Wpi, bpi, wmu_f, wsig_f, bmu, bsig,
        pi_new, mu_new, sigma_new);

    dens2_gamma_kernel<<<NPIX / 256, 256, 0, stream>>>(
        x, mu_new, sigma_new, pi_new, Wg, bg, gamma);
}
